// TravelingObserverModel_42030549958779
// MI455X (gfx1250) — matrix operand order repaired and checked by reading, not compiled, not run on hardware
//
#include <hip/hip_runtime.h>

typedef __attribute__((ext_vector_type(16))) __bf16 v16bf;
typedef __attribute__((ext_vector_type(8)))  float  v8f;

// ---------------------------------------------------------------------------
// Helpers
// ---------------------------------------------------------------------------
template <typename T>
__device__ __forceinline__ __bf16 to_bf16(T v) { return (__bf16)(float)v; }

// ---------------------------------------------------------------------------
// Generic NT GEMM on WMMA (one 16x16 tile per wave). Used only for the small,
// batch-independent GEMMs (scores / attn / FiLM maps).
//   Cout[m,n] = epilogue( sum_k A[m,k] * B[n,k] )
//   BIAS_MODE: 0 none, 1 bias[col], 2 bias[row]
// Requires M%16==0, N%16==0, K%32==0.
// ---------------------------------------------------------------------------
template <typename AT, typename OT, int BIAS_MODE>
__global__ void gemm_nt_wmma(const AT* __restrict__ A, const float* __restrict__ B,
                             OT* __restrict__ Cout,
                             const float* __restrict__ bias,
                             int M, int N, int K)
{
    const int lane = threadIdx.x & 31;
    const int wave = threadIdx.x >> 5;
    const int m0 = (blockIdx.x * 8 + wave) * 16;
    const int n0 = blockIdx.y * 16;
    if (m0 >= M) return;                           // uniform per wave

    const int half = lane >> 4;
    const int l15  = lane & 15;
    const AT*    abase = A + (size_t)(m0 + l15) * K;
    const float* bbase = B + (size_t)(n0 + l15) * K;

    v8f acc = {};
    for (int kc = 0; kc < K; kc += 32) {
        v16bf afrag, bfrag;
        const AT* arow = abase + kc + half * 8;
#pragma unroll
        for (int t = 0; t < 8; ++t) {
            afrag[t]     = to_bf16(arow[t]);
            afrag[t + 8] = to_bf16(arow[16 + t]);
        }
        const float* brow = bbase + kc + half * 8;
#pragma unroll
        for (int t = 0; t < 8; ++t) {
            bfrag[t]     = (__bf16)brow[t];
            bfrag[t + 8] = (__bf16)brow[16 + t];
        }

        if (kc + 32 < K) __builtin_prefetch(arow + 32, 0, 1);  // global_prefetch_b8
        acc = __builtin_amdgcn_wmma_f32_16x16x32_bf16(
            false, afrag, false, bfrag, (short)0, acc, false, false);
    }

    const int nB = n0 + l15;
    float bcol = 0.0f;
    if (BIAS_MODE == 1) bcol = bias[nB];
#pragma unroll
    for (int v = 0; v < 8; ++v) {
        const int mv = m0 + v + 8 * half;
        float val = acc[v];
        if (BIAS_MODE == 1) val += bcol;
        else if (BIAS_MODE == 2) val += bias[mv];
        Cout[(size_t)mv * N + nB] = (OT)val;
    }
}

// ---------------------------------------------------------------------------
// N-blocked NT GEMM for the heavy film/core layers (N = NTILES*16, N*K <= 16384).
// Block = 256 threads = 8 waves. B (N x K f32) is staged once into LDS as bf16;
// each wave owns 16 rows x full N: the A fragment is loaded ONCE per k-step
// and reused across NTILES WMMAs (8x less A traffic vs. per-tile blocking).
//   FILM:  val = g[n*nseq + (m%nseq)]*val + be[...]
//   RESID: val += resid[m*N+n]
// ---------------------------------------------------------------------------
template <typename AT, typename OT, int NTILES, int BIAS_MODE, bool FILM, bool RESID, bool RELU>
__global__ void gemm_nt_wmma_nb(const AT* __restrict__ A, const float* __restrict__ B,
                                OT* __restrict__ Cout,
                                const float* __restrict__ bias,
                                const float* __restrict__ gmap,
                                const float* __restrict__ bemap,
                                const OT* __restrict__ resid,
                                int M, int K, int nseq)
{
    constexpr int N = NTILES * 16;
    __shared__ __bf16 Blds[16384];                 // 32 KB of the 320 KB WGP LDS

    // cooperative stage: B (N x K, f32) -> LDS bf16, same row-major layout
    const int total = N * K;
    for (int i = threadIdx.x; i < total; i += 256) Blds[i] = (__bf16)B[i];
    __syncthreads();

    const int lane = threadIdx.x & 31;
    const int wave = threadIdx.x >> 5;
    const int m0 = (blockIdx.x * 8 + wave) * 16;
    if (m0 >= M) return;                           // uniform per wave

    const int half = lane >> 4;
    const int l15  = lane & 15;
    const AT* abase = A + (size_t)(m0 + l15) * K;

    v8f acc[NTILES];
    {
        v8f z = {};
#pragma unroll
        for (int nt = 0; nt < NTILES; ++nt) acc[nt] = z;
    }

    for (int kc = 0; kc < K; kc += 32) {
        v16bf afrag;
        const AT* arow = abase + kc + half * 16;
#pragma unroll
        for (int t = 0; t < 16; ++t) afrag[t] = to_bf16(arow[t]);
        if (kc + 32 < K) __builtin_prefetch(arow + 32, 0, 1);  // global_prefetch_b8
#pragma unroll
        for (int nt = 0; nt < NTILES; ++nt) {
            // 32B-aligned v16bf fragment straight from LDS (ds_load_b128 pair)
            const v16bf bfrag =
                *(const v16bf*)&Blds[(nt * 16 + l15) * K + kc + half * 16];
            acc[nt] = __builtin_amdgcn_wmma_f32_16x16x32_bf16(
                false, afrag, false, bfrag, (short)0, acc[nt], false, false);
        }
    }

#pragma unroll
    for (int nt = 0; nt < NTILES; ++nt) {
        const int nB = nt * 16 + l15;
        float bcol = 0.0f;
        if (BIAS_MODE == 1) bcol = bias[nB];
#pragma unroll
        for (int v = 0; v < 8; ++v) {
            const int mv = m0 + v + 8 * half;
            float val = acc[nt][v];
            if (BIAS_MODE == 1) val += bcol;
            else if (BIAS_MODE == 2) val += bias[mv];
            if (FILM) {
                const int npos = mv % nseq;
                const size_t gi = (size_t)nB * nseq + npos;
                val = gmap[gi] * val + bemap[gi];
            }
            if (RESID) val += (float)resid[(size_t)mv * N + nB];
            if (RELU) val = fmaxf(val, 0.0f);
            Cout[(size_t)mv * N + nB] = (OT)val;
        }
    }
}

// ---------------------------------------------------------------------------
// Context gather: out[i, c] = dataset_ctx[ var_indices[ sel[i] ], c ]
// ---------------------------------------------------------------------------
__global__ void gather_ctx_kernel(const int* __restrict__ var_indices,
                                  const int* __restrict__ sel,
                                  const float* __restrict__ dataset_ctx,
                                  float* __restrict__ out, int Nsel, int C)
{
    int i = blockIdx.x * blockDim.x + threadIdx.x;
    if (i >= Nsel * C) return;
    int row = i / C, c = i % C;
    int v = var_indices[sel[row]];
    out[i] = dataset_ctx[(size_t)v * C + c];
}

// basisT[c, s] = basis[s, c]
__global__ void transpose_kernel(const float* __restrict__ in, float* __restrict__ out,
                                 int R, int Ccols)
{
    int i = blockIdx.x * blockDim.x + threadIdx.x;
    if (i >= R * Ccols) return;
    int r = i / Ccols, c = i % Ccols;
    out[(size_t)c * R + r] = in[i];
}

// ---------------------------------------------------------------------------
// alpha-entmax over each row (length S) of `scores`, in place, via bisection.
// One 256-thread workgroup per row; row cached in registers (S/256 <= 8).
// ---------------------------------------------------------------------------
__global__ void entmax_kernel(float* __restrict__ scores,
                              const float* __restrict__ alpha_p, int S)
{
    __shared__ float red[256];
    const int t = threadIdx.x;
    float* row = scores + (size_t)blockIdx.x * S;
    const int ept = S / 256;
    const float am1 = alpha_p[0] - 1.0f;
    const float inv = 1.0f / am1;

    float xs[8];
    float m = -3.4e38f;
    for (int j = 0; j < ept; ++j) {
        float v = row[t + j * 256] * am1;
        xs[j] = v;
        m = fmaxf(m, v);
    }
    red[t] = m; __syncthreads();
    for (int o = 128; o > 0; o >>= 1) { if (t < o) red[t] = fmaxf(red[t], red[t + o]); __syncthreads(); }
    const float maxv = red[0]; __syncthreads();

    float tau = maxv - 1.0f;
    float dm  = (maxv - __powf(1.0f / (float)S, am1)) - tau;

    for (int it = 0; it < 50; ++it) {
        dm *= 0.5f;
        const float tm = tau + dm;
        float s = 0.0f;
        for (int j = 0; j < ept; ++j) {
            float d = xs[j] - tm;
            if (d > 0.0f) s += __powf(d, inv);
        }
        red[t] = s; __syncthreads();
        for (int o = 128; o > 0; o >>= 1) { if (t < o) red[t] += red[t + o]; __syncthreads(); }
        const float f = red[0] - 1.0f; __syncthreads();
        if (f >= 0.0f) tau = tm;       // uniform: red[0] identical across block
    }

    float pv[8], s = 0.0f;
    for (int j = 0; j < ept; ++j) {
        float d = xs[j] - tau;
        pv[j] = d > 0.0f ? __powf(d, inv) : 0.0f;
        s += pv[j];
    }
    red[t] = s; __syncthreads();
    for (int o = 128; o > 0; o >>= 1) { if (t < o) red[t] += red[t + o]; __syncthreads(); }
    const float tot = red[0];
    for (int j = 0; j < ept; ++j) row[t + j * 256] = pv[j] / tot;
}

// ---------------------------------------------------------------------------
// Input FiLM layer (encf): x[b,n,o] = relu(g[o,n]*(W[o]*in[b,n] + b[o]) + be[o,n])
// ---------------------------------------------------------------------------
__global__ void encf_kernel(const float* __restrict__ input,
                            const float* __restrict__ W, const float* __restrict__ bias,
                            const float* __restrict__ g, const float* __restrict__ be,
                            __bf16* __restrict__ x, int B, int Nin, int H)
{
    int i = blockIdx.x * blockDim.x + threadIdx.x;
    if (i >= B * Nin * H) return;
    int o = i % H;
    int n = (i / H) % Nin;
    int b = i / (H * Nin);
    float h = W[o] * input[(size_t)b * Nin + n] + bias[o];
    float v = g[(size_t)o * Nin + n] * h + be[(size_t)o * Nin + n];
    x[i] = (__bf16)fmaxf(v, 0.0f);
}

// s[b,o] = sum_n x[b,n,o]
__global__ void nsum_kernel(const __bf16* __restrict__ x, float* __restrict__ s,
                            int Nseq, int H)
{
    const int b = blockIdx.x, o = threadIdx.x;
    const __bf16* p = x + (size_t)b * Nseq * H + o;
    float acc = 0.0f;
    for (int n = 0; n < Nseq; ++n) acc += (float)p[(size_t)n * H];
    s[(size_t)b * H + o] = acc;
}

// xdec[b,n,o] = bf16(s[b,o])
__global__ void broadcast_kernel(const float* __restrict__ s, __bf16* __restrict__ x,
                                 int B, int Nout, int H)
{
    int i = blockIdx.x * blockDim.x + threadIdx.x;
    if (i >= B * Nout * H) return;
    int o = i % H;
    int b = i / (H * Nout);
    x[i] = (__bf16)s[(size_t)b * H + o];
}

// Final FiLM (decf): y[b,n] = g(n)*(decf_W . x[b,n,:] + decf_b) + be(n)
__global__ void decf_kernel(const __bf16* __restrict__ x, const float* __restrict__ out_ctx,
                            const float* __restrict__ W, const float* __restrict__ b0,
                            const float* __restrict__ Gw, const float* __restrict__ Gb,
                            const float* __restrict__ Bw, const float* __restrict__ Bb,
                            float* __restrict__ y, int B, int Nout, int H, int C)
{
    int i = blockIdx.x * blockDim.x + threadIdx.x;
    if (i >= B * Nout) return;
    int n = i % Nout;
    float h = b0[0];
    const __bf16* xr = x + (size_t)i * H;
    for (int o = 0; o < H; ++o) h += W[o] * (float)xr[o];
    float g = Gb[0], be = Bb[0];
    const float* ctx = out_ctx + (size_t)n * C;
    for (int c = 0; c < C; ++c) { g += Gw[c] * ctx[c]; be += Bw[c] * ctx[c]; }
    y[i] = g * h + be;
}

// ---------------------------------------------------------------------------
// Host orchestration
// ---------------------------------------------------------------------------
extern "C" void kernel_launch(void* const* d_in, const int* in_sizes, int n_in,
                              void* d_out, int out_size, void* d_ws, size_t ws_size,
                              hipStream_t stream)
{
    (void)n_in; (void)out_size; (void)ws_size;

    const float* input_batch = (const float*)d_in[0];
    const int*   var_indices = (const int*)d_in[1];
    const int*   in_var_idx  = (const int*)d_in[2];
    const int*   out_var_idx = (const int*)d_in[3];
    const float* dataset_ctx = (const float*)d_in[4];
    const float* basis_ctx   = (const float*)d_in[5];
    const float* alpha_p     = (const float*)d_in[6];
    const float* encf_W  = (const float*)d_in[7];
    const float* encf_b  = (const float*)d_in[8];
    const float* encf_Gw = (const float*)d_in[9];
    const float* encf_Gb = (const float*)d_in[10];
    const float* encf_Bw = (const float*)d_in[11];
    const float* encf_Bb = (const float*)d_in[12];
    const float* enc_W   = (const float*)d_in[13];
    const float* enc_b   = (const float*)d_in[14];
    const float* enc_Gw  = (const float*)d_in[15];
    const float* enc_Gb  = (const float*)d_in[16];
    const float* enc_Bw  = (const float*)d_in[17];
    const float* enc_Bb  = (const float*)d_in[18];
    const float* core_W  = (const float*)d_in[19];
    const float* core_b  = (const float*)d_in[20];
    const float* dec_W   = (const float*)d_in[21];
    const float* dec_b   = (const float*)d_in[22];
    const float* dec_Gw  = (const float*)d_in[23];
    const float* dec_Gb  = (const float*)d_in[24];
    const float* dec_Bw  = (const float*)d_in[25];
    const float* dec_Bb  = (const float*)d_in[26];
    const float* decf_W  = (const float*)d_in[27];
    const float* decf_b  = (const float*)d_in[28];
    const float* decf_Gw = (const float*)d_in[29];
    const float* decf_Gb = (const float*)d_in[30];
    const float* decf_Bw = (const float*)d_in[31];
    const float* decf_Bb = (const float*)d_in[32];

    const int H    = in_sizes[7];             // encf_W: (H,1)
    const int C    = in_sizes[9] / H;         // encf_Gw: (H,C)
    const int Nin  = in_sizes[2];
    const int Nout = in_sizes[3];
    const int B    = in_sizes[0] / Nin;
    const int S    = in_sizes[5] / C;
    const int Lenc  = in_sizes[13] / (2 * H * H);
    const int Lcore = in_sizes[19] / (2 * H * H);
    const int Ldec  = in_sizes[21] / (2 * H * H);

    // ---- workspace carve-up -------------------------------------------------
    char* ws = (char*)d_ws;
    size_t off = 0;
    auto alloc = [&](size_t bytes) -> char* {
        char* p = ws + off;
        off = (off + bytes + 255) & ~(size_t)255;
        return p;
    };
    float*  in_ctx  = (float*)alloc((size_t)Nin * C * 4);
    float*  out_ctx = (float*)alloc((size_t)Nout * C * 4);
    float*  basisT  = (float*)alloc((size_t)C * S * 4);
    float*  scores  = (float*)alloc((size_t)Nin * S * 4);   // becomes p in place
    float*  attn    = (float*)alloc((size_t)Nin * C * 4);
    const int nEncMaps = 1 + Lenc * 2;                      // [0]=encf, then enc[l][s]
    float*  enc_g  = (float*)alloc((size_t)nEncMaps * H * Nin * 4);
    float*  enc_beM= (float*)alloc((size_t)nEncMaps * H * Nin * 4);
    const int nDecMaps = Ldec * 2;
    float*  dec_g  = (float*)alloc((size_t)nDecMaps * H * Nout * 4);
    float*  dec_beM= (float*)alloc((size_t)nDecMaps * H * Nout * 4);
    __bf16* xenc   = (__bf16*)alloc((size_t)B * Nin * H * 2);
    __bf16* henc   = (__bf16*)alloc((size_t)B * Nin * H * 2);
    float*  s_core = (float*)alloc((size_t)B * H * 4);
    float*  h_core = (float*)alloc((size_t)B * H * 4);
    __bf16* xdec   = (__bf16*)alloc((size_t)B * Nout * H * 2);
    __bf16* hdec   = (__bf16*)alloc((size_t)B * Nout * H * 2);

    const dim3 blk(256);
    auto gemmGrid = [](int M, int N) { return dim3((unsigned)((M + 127) / 128), (unsigned)(N / 16)); };
    auto nbGrid   = [](int M)        { return dim3((unsigned)((M + 127) / 128)); };

    // ---- batch-independent attention head ----------------------------------
    {
        int tot = Nin * C;
        gather_ctx_kernel<<<(tot + 255) / 256, blk, 0, stream>>>(var_indices, in_var_idx, dataset_ctx, in_ctx, Nin, C);
        tot = Nout * C;
        gather_ctx_kernel<<<(tot + 255) / 256, blk, 0, stream>>>(var_indices, out_var_idx, dataset_ctx, out_ctx, Nout, C);
        tot = S * C;
        transpose_kernel<<<(tot + 255) / 256, blk, 0, stream>>>(basis_ctx, basisT, S, C);
    }
    // scores[n,s] = in_ctx . basis_ctx   (M=Nin, N=S, K=C)
    gemm_nt_wmma<float, float, 0><<<gemmGrid(Nin, S), blk, 0, stream>>>(
        in_ctx, basis_ctx, scores, nullptr, Nin, S, C);
    // entmax rows, in place (scores -> p)
    entmax_kernel<<<Nin, blk, 0, stream>>>(scores, alpha_p, S);
    // attn[n,c] = p . basisT   (M=Nin, N=C, K=S)
    gemm_nt_wmma<float, float, 0><<<gemmGrid(Nin, C), blk, 0, stream>>>(
        scores, basisT, attn, nullptr, Nin, C, S);

    // ---- FiLM gate/bias maps (batch-independent) ---------------------------
    // g[o,n] = Gw . attn[n,:] + Gb[o]   (M=H, N=Nseq, K=C, bias per row)
    auto launch_map = [&](const float* Wc, const float* bc, const float* ctx,
                          float* dst, int Nseq) {
        gemm_nt_wmma<float, float, 2><<<gemmGrid(H, Nseq), blk, 0, stream>>>(
            Wc, ctx, dst, bc, H, Nseq, C);
    };
    launch_map(encf_Gw, encf_Gb, attn, enc_g, Nin);
    launch_map(encf_Bw, encf_Bb, attn, enc_beM, Nin);
    for (int l = 0; l < Lenc; ++l)
        for (int s1 = 0; s1 < 2; ++s1) {
            const int li = l * 2 + s1, mi = 1 + li;
            launch_map(enc_Gw + (size_t)li * H * C, enc_Gb + (size_t)li * H, attn,
                       enc_g + (size_t)mi * H * Nin, Nin);
            launch_map(enc_Bw + (size_t)li * H * C, enc_Bb + (size_t)li * H, attn,
                       enc_beM + (size_t)mi * H * Nin, Nin);
        }
    for (int l = 0; l < Ldec; ++l)
        for (int s1 = 0; s1 < 2; ++s1) {
            const int li = l * 2 + s1;
            launch_map(dec_Gw + (size_t)li * H * C, dec_Gb + (size_t)li * H, out_ctx,
                       dec_g + (size_t)li * H * Nout, Nout);
            launch_map(dec_Bw + (size_t)li * H * C, dec_Bb + (size_t)li * H, out_ctx,
                       dec_beM + (size_t)li * H * Nout, Nout);
        }

    // ---- encoder ------------------------------------------------------------
    {
        int tot = B * Nin * H;
        encf_kernel<<<(tot + 255) / 256, blk, 0, stream>>>(input_batch, encf_W, encf_b,
                                                           enc_g, enc_beM, xenc, B, Nin, H);
    }
    const int Menc = B * Nin;
    for (int l = 0; l < Lenc; ++l) {
        const int li0 = l * 2, li1 = l * 2 + 1;
        // h = relu(g*(x W0^T + b0) + be)
        gemm_nt_wmma_nb<__bf16, __bf16, 8, 1, true, false, true><<<nbGrid(Menc), blk, 0, stream>>>(
            xenc, enc_W + (size_t)li0 * H * H, henc, enc_b + (size_t)li0 * H,
            enc_g + (size_t)(1 + li0) * H * Nin, enc_beM + (size_t)(1 + li0) * H * Nin,
            nullptr, Menc, H, Nin);
        // x = relu(x + g*(h W1^T + b1) + be)
        gemm_nt_wmma_nb<__bf16, __bf16, 8, 1, true, true, true><<<nbGrid(Menc), blk, 0, stream>>>(
            henc, enc_W + (size_t)li1 * H * H, xenc, enc_b + (size_t)li1 * H,
            enc_g + (size_t)(1 + li1) * H * Nin, enc_beM + (size_t)(1 + li1) * H * Nin,
            xenc, Menc, H, Nin);
    }

    // ---- reduce over n, core MLP -------------------------------------------
    nsum_kernel<<<B, dim3((unsigned)H), 0, stream>>>(xenc, s_core, Nin, H);
    for (int l = 0; l < Lcore; ++l) {
        const int li0 = l * 2, li1 = l * 2 + 1;
        gemm_nt_wmma_nb<float, float, 8, 1, false, false, true><<<nbGrid(B), blk, 0, stream>>>(
            s_core, core_W + (size_t)li0 * H * H, h_core, core_b + (size_t)li0 * H,
            nullptr, nullptr, nullptr, B, H, 1);
        gemm_nt_wmma_nb<float, float, 8, 1, false, true, true><<<nbGrid(B), blk, 0, stream>>>(
            h_core, core_W + (size_t)li1 * H * H, s_core, core_b + (size_t)li1 * H,
            nullptr, nullptr, s_core, B, H, 1);
    }

    // ---- decoder ------------------------------------------------------------
    {
        int tot = B * Nout * H;
        broadcast_kernel<<<(tot + 255) / 256, blk, 0, stream>>>(s_core, xdec, B, Nout, H);
    }
    const int Mdec = B * Nout;
    for (int l = 0; l < Ldec; ++l) {
        const int li0 = l * 2, li1 = l * 2 + 1;
        gemm_nt_wmma_nb<__bf16, __bf16, 8, 1, true, false, true><<<nbGrid(Mdec), blk, 0, stream>>>(
            xdec, dec_W + (size_t)li0 * H * H, hdec, dec_b + (size_t)li0 * H,
            dec_g + (size_t)li0 * H * Nout, dec_beM + (size_t)li0 * H * Nout,
            nullptr, Mdec, H, Nout);
        gemm_nt_wmma_nb<__bf16, __bf16, 8, 1, true, true, true><<<nbGrid(Mdec), blk, 0, stream>>>(
            hdec, dec_W + (size_t)li1 * H * H, xdec, dec_b + (size_t)li1 * H,
            dec_g + (size_t)li1 * H * Nout, dec_beM + (size_t)li1 * H * Nout,
            xdec, Mdec, H, Nout);
    }

    // ---- final FiLM head ----------------------------------------------------
    {
        int tot = B * Nout;
        decf_kernel<<<(tot + 255) / 256, blk, 0, stream>>>(xdec, out_ctx, decf_W, decf_b,
                                                           decf_Gw, decf_Gb, decf_Bw, decf_Bb,
                                                           (float*)d_out, B, Nout, H, C);
    }
}
